// MambaLayer_29910152249595
// MI455X (gfx1250) — matrix operand order repaired and checked by reading, not compiled, not run on hardware
//
#include <hip/hip_runtime.h>
#include <hip/hip_bf16.h>

// ---------------- problem constants (match reference) ----------------
#define BB      2
#define DMODEL  128
#define LSEQ    8192          // DZ*HH*WW = 8*32*32
#define DINNER  256
#define DSTATE  16
#define DTRANK  8
#define DCONV   4
#define NXZ     (2*DINNER)    // 512
#define NDBL    (DTRANK + 2*DSTATE)  // 40
#define MTOT    (BB*LSEQ)     // 16384

typedef __attribute__((ext_vector_type(16))) _Float16 v16h;
typedef __attribute__((ext_vector_type(8)))  _Float16 v8h;
typedef __attribute__((ext_vector_type(8)))  float    v8f;

// ---------------- WMMA fragment loads (CDNA5 ISA 7.12.2 layouts) ----------------
// A 16x32 f16 (M x K): lane m = lane&15; halves 0..7 = K (kc + (laneHi?8:0) + i),
// halves 8..15 = K (kc+16 + (laneHi?8:0) + i). Two contiguous 16B runs per lane.
__device__ __forceinline__ v16h load_a_frag(const _Float16* __restrict__ A, int K,
                                            int m0, int kc, int lane) {
    int m   = m0 + (lane & 15);
    int off = (lane & 16) ? 8 : 0;
    const _Float16* p = A + (size_t)m * K + kc + off;
    v8h lo = *(const v8h*)(p);
    v8h hi = *(const v8h*)(p + 16);
    v16h r;
#pragma unroll
    for (int i = 0; i < 8; ++i) { r[i] = lo[i]; r[i + 8] = hi[i]; }
    return r;
}

// B 32x16 f16 (K x N), stored K-major (Bt[k*N + n]): lane n = lane&15; halves 0..7 = K (kc + (laneHi?8:0) + i),
// halves 8..15 = K (kc+16 + (laneHi?8:0) + i).
__device__ __forceinline__ v16h load_b_frag(const _Float16* __restrict__ Bt, int N,
                                            int n0, int kc, int lane) {
    int n   = n0 + (lane & 15);
    int off = (lane & 16) ? 8 : 0;
    const _Float16* p = Bt + (size_t)(kc + off) * N + n;
    v16h r;
#pragma unroll
    for (int i = 0; i < 8; ++i) { r[i] = p[(size_t)i * N]; r[i + 8] = p[(size_t)(i + 16) * N]; }
    return r;
}

// ---------------- generic WMMA GEMM: C[M,N] = A[M,K(f16)] x Bt[K,N(f16)] ----------------
// TOUT=0: row-major C.  TOUT=1: scatter C[(b*N + n)*Lout + l] with m = b*Lout + l.
template <int TOUT>
__global__ void wmma_gemm_kernel(const _Float16* __restrict__ A,
                                 const _Float16* __restrict__ Bt,
                                 float* __restrict__ C,
                                 int M, int N, int K, int Lout) {
    int lane = threadIdx.x;                          // 0..31 (wave32)
    int n0 = blockIdx.x * 16;
    int m0 = (blockIdx.y * blockDim.y + threadIdx.y) * 16;
    if (m0 >= M || n0 >= N) return;                  // wave-uniform

    v8f acc = {};
    for (int kc = 0; kc < K; kc += 32) {
        v16h a = load_a_frag(A, K, m0, kc, lane);
        v16h b = load_b_frag(Bt, N, n0, kc, lane);
        acc = __builtin_amdgcn_wmma_f32_16x16x32_f16(false, a, false, b,
                                                     (short)0, acc, false, false);
    }
    // C/D layout: VGPR r -> M = m0 + r + (laneHi?8:0), N = n0 + lane&15
    int nn = n0 + (lane & 15);
    int mb = m0 + ((lane & 16) ? 8 : 0);
#pragma unroll
    for (int r = 0; r < 8; ++r) {
        int m = mb + r;
        if (TOUT == 0) {
            C[(size_t)m * N + nn] = acc[r];
        } else {
            int b = m / Lout, l = m % Lout;
            C[((size_t)b * N + nn) * Lout + l] = acc[r];
        }
    }
}

// ---------------- weight prep: f32 -> f16, transposed to K-major ----------------
__global__ void prep_weights_kernel(const float* __restrict__ W_in,   // (512,128)
                                    const float* __restrict__ W_out,  // (128,256)
                                    _Float16* __restrict__ Wt_in,     // (128,512) k-major
                                    _Float16* __restrict__ Wt_out) {  // (256,128) k-major
    int i = blockIdx.x * 256 + threadIdx.x;
    if (i < NXZ * DMODEL) {
        int n = i / DMODEL, k = i % DMODEL;
        Wt_in[k * NXZ + n] = (_Float16)W_in[i];
    }
    if (i < DMODEL * DINNER) {
        int e = i / DINNER, k = i % DINNER;
        Wt_out[k * DMODEL + e] = (_Float16)W_out[i];
    }
}

// ---------------- LayerNorm over D_MODEL; input x is (B, D, L); one wave per token --------
__global__ void ln_kernel(const float* __restrict__ x,
                          const float* __restrict__ g,
                          const float* __restrict__ beta,
                          _Float16* __restrict__ xn) {
    int lane = threadIdx.x & 31;
    int row  = blockIdx.x * 8 + (threadIdx.x >> 5);   // b*L + l
    int b = row / LSEQ, l = row % LSEQ;
    const float* xb = x + (size_t)b * DMODEL * LSEQ + l;
    float v[4];
    float s = 0.f;
#pragma unroll
    for (int j = 0; j < 4; ++j) { v[j] = xb[(size_t)(lane + 32 * j) * LSEQ]; s += v[j]; }
#pragma unroll
    for (int msk = 16; msk >= 1; msk >>= 1) s += __shfl_xor(s, msk);
    float mu = s * (1.f / DMODEL);
    float q = 0.f;
#pragma unroll
    for (int j = 0; j < 4; ++j) { float t = v[j] - mu; q += t * t; }
#pragma unroll
    for (int msk = 16; msk >= 1; msk >>= 1) q += __shfl_xor(q, msk);
    float rs = rsqrtf(q * (1.f / DMODEL) + 1e-5f);
#pragma unroll
    for (int j = 0; j < 4; ++j) {
        int d = lane + 32 * j;
        float o = (v[j] - mu) * rs * g[d] + beta[d];
        xn[(size_t)row * DMODEL + d] = (_Float16)o;
    }
}

// ---------------- depthwise causal conv (k=4) + SiLU ----------------
__global__ void conv_silu_kernel(const float* __restrict__ xz,      // (M,512): [:256]=xm_raw
                                 const float* __restrict__ conv_w,  // (256,1,4)
                                 const float* __restrict__ conv_b,
                                 float* __restrict__ xm_f,
                                 _Float16* __restrict__ xm_h) {
    size_t idx = (size_t)blockIdx.x * 256 + threadIdx.x;   // (b*L + l)*256 + d
    int d = (int)(idx & (DINNER - 1));
    size_t m = idx >> 8;
    int l = (int)(m % LSEQ);
    float acc = conv_b[d];
#pragma unroll
    for (int j = 0; j < DCONV; ++j) {
        int ll = l - (DCONV - 1) + j;
        if (ll >= 0) acc += conv_w[d * DCONV + j] * xz[(m - (DCONV - 1) + j) * NXZ + d];
    }
    float sv = acc / (1.f + __expf(-acc));  // silu
    xm_f[m * DINNER + d] = sv;
    xm_h[m * DINNER + d] = (_Float16)sv;
}

// ---------------- x_dbl = xm @ W_xproj^T  (N=40 too skinny for WMMA) ----------------
__global__ void xproj_kernel(const float* __restrict__ xm_f,
                             const float* __restrict__ W_xproj,   // (40,256)
                             float* __restrict__ xdbl) {
    size_t idx = (size_t)blockIdx.x * 256 + threadIdx.x;     // m*40 + j
    int j = (int)(idx % NDBL);
    size_t m = idx / NDBL;
    const float* xr = xm_f + m * DINNER;
    const float* wr = W_xproj + (size_t)j * DINNER;
    float acc = 0.f;
#pragma unroll 8
    for (int d = 0; d < DINNER; ++d) acc += xr[d] * wr[d];
    xdbl[m * NDBL + j] = acc;
}

// ---------------- delta = softplus(dt_raw @ W_dt^T + b_dt) ----------------
__global__ void delta_kernel(const float* __restrict__ xdbl,
                             const float* __restrict__ W_dt,    // (256,8)
                             const float* __restrict__ b_dt,
                             float* __restrict__ delta) {
    size_t idx = (size_t)blockIdx.x * 256 + threadIdx.x;      // m*256 + d
    int d = (int)(idx & (DINNER - 1));
    size_t m = idx >> 8;
    float acc = b_dt[d];
#pragma unroll
    for (int r = 0; r < DTRANK; ++r) acc += xdbl[m * NDBL + r] * W_dt[d * DTRANK + r];
    float sp = (acc > 20.f) ? acc : log1pf(__expf(acc));
    delta[m * DINNER + d] = sp;
}

// ---------------- selective scan: lane = state n, 16 lanes per (b,d) channel ----------
__global__ void scan_kernel(const float* __restrict__ delta,
                            const float* __restrict__ xdbl,
                            const float* __restrict__ xm_f,
                            const float* __restrict__ A_log,
                            float* __restrict__ ycore) {
    int lane = threadIdx.x & 31;
    int wave = threadIdx.x >> 5;                          // 0..7
    int ch = blockIdx.x * 16 + wave * 2 + (lane >> 4);    // 0..511 = b*256 + d
    int b = ch >> 8, d = ch & (DINNER - 1);
    int n = lane & (DSTATE - 1);
    float Adn = -__expf(A_log[d * DSTATE + n]);
    float h = 0.f;
    size_t base = (size_t)b * LSEQ;
    for (int l = 0; l < LSEQ; ++l) {
        size_t m = base + l;
        float dv = delta[m * DINNER + d];
        float xv = xm_f[m * DINNER + d];
        float Bv = xdbl[m * NDBL + DTRANK + n];
        float Cv = xdbl[m * NDBL + DTRANK + DSTATE + n];
        h = h * __expf(dv * Adn) + dv * Bv * xv;
        float c = h * Cv;
        c += __shfl_xor(c, 1);
        c += __shfl_xor(c, 2);
        c += __shfl_xor(c, 4);
        c += __shfl_xor(c, 8);
        if (n == 0) ycore[m * DINNER + d] = c;
    }
}

// ---------------- y = (ycore + xm*D) * silu(z), convert f16 for out GEMM ----------
__global__ void gate_kernel(const float* __restrict__ ycore,
                            const float* __restrict__ xm_f,
                            const float* __restrict__ xz,
                            const float* __restrict__ Dp,
                            _Float16* __restrict__ y_h) {
    size_t idx = (size_t)blockIdx.x * 256 + threadIdx.x;   // m*256 + d
    int d = (int)(idx & (DINNER - 1));
    size_t m = idx >> 8;
    float zv = xz[m * NXZ + DINNER + d];
    float sz = zv / (1.f + __expf(-zv));
    float y = (ycore[m * DINNER + d] + xm_f[m * DINNER + d] * Dp[d]) * sz;
    y_h[m * DINNER + d] = (_Float16)y;
}

// ---------------- host-side launcher ----------------
extern "C" void kernel_launch(void* const* d_in, const int* in_sizes, int n_in,
                              void* d_out, int out_size, void* d_ws, size_t ws_size,
                              hipStream_t stream) {
    const float* x        = (const float*)d_in[0];
    const float* gamma    = (const float*)d_in[1];
    const float* beta     = (const float*)d_in[2];
    const float* W_in     = (const float*)d_in[3];
    const float* conv_w   = (const float*)d_in[4];
    const float* conv_b   = (const float*)d_in[5];
    const float* W_xproj  = (const float*)d_in[6];
    const float* W_dt     = (const float*)d_in[7];
    const float* b_dt     = (const float*)d_in[8];
    const float* A_log    = (const float*)d_in[9];
    const float* D_param  = (const float*)d_in[10];
    const float* W_out    = (const float*)d_in[11];
    float* out = (float*)d_out;

    // workspace carve-up (all 256B aligned)
    char* ws = (char*)d_ws;
    size_t cur = 0;
    auto alloc = [&](size_t bytes) -> char* {
        char* p = ws + cur;
        cur += (bytes + 255) & ~(size_t)255;
        return p;
    };
    _Float16* xn_h   = (_Float16*)alloc((size_t)MTOT * DMODEL * 2);   //  4 MB
    _Float16* Wt_in  = (_Float16*)alloc((size_t)DMODEL * NXZ * 2);    // 128 KB
    _Float16* Wt_out = (_Float16*)alloc((size_t)DINNER * DMODEL * 2); //  64 KB
    float*    xz     = (float*)   alloc((size_t)MTOT * NXZ * 4);      // 32 MB
    float*    xm_f   = (float*)   alloc((size_t)MTOT * DINNER * 4);   // 16 MB
    _Float16* xm_h   = (_Float16*)alloc((size_t)MTOT * DINNER * 2);   //  8 MB
    float*    xdbl   = (float*)   alloc((size_t)MTOT * NDBL * 4);     // 2.5 MB
    float*    delta  = (float*)   alloc((size_t)MTOT * DINNER * 4);   // 16 MB
    float*    ycore  = (float*)   alloc((size_t)MTOT * DINNER * 4);   // 16 MB
    _Float16* y_h    = (_Float16*)alloc((size_t)MTOT * DINNER * 2);   //  8 MB
    (void)ws_size; (void)in_sizes; (void)n_in; (void)out_size;

    // 1. weight prep (f32 -> f16, K-major transpose)
    prep_weights_kernel<<<(NXZ * DMODEL + 255) / 256, 256, 0, stream>>>(
        W_in, W_out, Wt_in, Wt_out);

    // 2. layernorm -> xn_h (f16), one wave per token
    ln_kernel<<<MTOT / 8, 256, 0, stream>>>(x, gamma, beta, xn_h);

    // 3. in-projection GEMM (WMMA): xz[16384,512] = xn[16384,128] x Wt_in[128,512]
    {
        dim3 blk(32, 8);
        dim3 grd(NXZ / 16, MTOT / 16 / 8);
        wmma_gemm_kernel<0><<<grd, blk, 0, stream>>>(xn_h, Wt_in, xz,
                                                     MTOT, NXZ, DMODEL, LSEQ);
    }

    // 4. depthwise causal conv + SiLU
    conv_silu_kernel<<<(size_t)MTOT * DINNER / 256, 256, 0, stream>>>(
        xz, conv_w, conv_b, xm_f, xm_h);

    // 5. x projection (dt/B/C)
    xproj_kernel<<<(size_t)MTOT * NDBL / 256, 256, 0, stream>>>(xm_f, W_xproj, xdbl);

    // 6. delta = softplus(...)
    delta_kernel<<<(size_t)MTOT * DINNER / 256, 256, 0, stream>>>(xdbl, W_dt, b_dt, delta);

    // 7. selective scan (recurrence over L)
    scan_kernel<<<(BB * DINNER) / 16, 256, 0, stream>>>(delta, xdbl, xm_f, A_log, ycore);

    // 8. gate: y = (ycore + xm*D) * silu(z)
    gate_kernel<<<(size_t)MTOT * DINNER / 256, 256, 0, stream>>>(ycore, xm_f, xz, D_param, y_h);

    // 9. out-projection GEMM (WMMA) with transposed scatter-store into (B, D_MODEL, L)
    {
        dim3 blk(32, 8);
        dim3 grd(DMODEL / 16, MTOT / 16 / 8);
        wmma_gemm_kernel<1><<<grd, blk, 0, stream>>>(y_h, Wt_out, out,
                                                     MTOT, DMODEL, DINNER, LSEQ);
    }
}
